// Attention_34832184770944
// MI455X (gfx1250) — compile-verified, matrix operand order checked by reading, not run on hardware
//
#include <hip/hip_runtime.h>

// ---------------------------------------------------------------------------
// GQA attention block for MI455X (gfx1250, wave32, WMMA bf16 + TDM).
// Pipeline: f32->bf16 cvt (weights pre-transposed) -> TDM-fed WMMA GEMM
//           (Q/K/V proj) -> RoPE+relayout -> WMMA flash attention (online
//           softmax) -> TDM-fed WMMA GEMM (wo).
// Workspace requirement: ~113 MB.
// ---------------------------------------------------------------------------

constexpr int kDim   = 2048;
constexpr int kNH    = 32;
constexpr int kNKV   = 8;
constexpr int kHD    = 64;
constexpr int kBsz   = 2;
constexpr int kSeq   = 2048;
constexpr int kRows  = kBsz * kSeq;    // 4096
constexpr int kKVD   = kNKV * kHD;     // 512

typedef __bf16 bf16;
typedef __attribute__((ext_vector_type(16))) __bf16 v16bf;
typedef __attribute__((ext_vector_type(8)))  __bf16 v8bf;
typedef __attribute__((ext_vector_type(4)))  __bf16 v4bf;
typedef __attribute__((ext_vector_type(8)))  float  v8f;
typedef __attribute__((ext_vector_type(4)))  float  v4f;
typedef __attribute__((ext_vector_type(4)))  unsigned v4u;
typedef __attribute__((ext_vector_type(8)))  unsigned v8u;

static __device__ __forceinline__ bf16 f2bf(float f) {
  unsigned u = __float_as_uint(f);
  u += 0x7FFFu + ((u >> 16) & 1u);          // round-to-nearest-even
  unsigned short h = (unsigned short)(u >> 16);
  return __builtin_bit_cast(bf16, h);
}

static __device__ __forceinline__ v8f v8f_zero() {
  v8f z;
#pragma unroll
  for (int i = 0; i < 8; ++i) z[i] = 0.0f;
  return z;
}

// Build a 16x32 bf16 A/B fragment register (v16bf) from two contiguous
// 16-byte chunks: elements 0..7 = K=kb..kb+7 (VGPR0-3), 8..15 = K=16+kb..
static __device__ __forceinline__ v16bf make_frag(const bf16* lo, const bf16* hi) {
  union { v16bf v; v8bf h[2]; } u;
  u.h[0] = *(const v8bf*)lo;
  u.h[1] = *(const v8bf*)hi;
  return u.v;
}

#define WMMA_BF16(A_, B_, C_) \
  __builtin_amdgcn_wmma_f32_16x16x32_bf16(false, (A_), false, (B_), (short)0, (C_), false, false)

// Low 32 bits of a generic pointer into LDS = byte offset within the
// wave's LDS allocation (ISA 10.2: LDS aperture keeps the offset in addr[31:0]).
static __device__ __forceinline__ unsigned lds_off(const void* p) {
  return (unsigned)(unsigned long long)p;
}

// ---------------------------------------------------------------------------
// Tensor Data Mover: async 2D tile load (tile 32 x 128 rows, 2-byte elems)
// from a row-major tensor (rows x rowLen) into LDS laid out [128][32].
// D# per CDNA5 ISA ch.8; issued by one wave, tracked with TENSORcnt.
// ---------------------------------------------------------------------------
static __device__ __forceinline__ void tdm_load_tile_2d(const bf16* gsrc,
                                                        unsigned ldsByteOff,
                                                        unsigned rowLen,
                                                        unsigned rows) {
  unsigned long long ga = (unsigned long long)gsrc;
  v4u g0;
  g0[0] = 1u;                                   // count=1, user descriptor
  g0[1] = ldsByteOff;                           // lds_addr
  g0[2] = (unsigned)ga;                         // global_addr[31:0]
  g0[3] = (unsigned)(ga >> 32) | (2u << 30);    // global_addr[56:32] | type=2
  v8u g1;
  g1[0] = 0x00010000u;                          // wg_mask=0, data_size=1 (2B)
  g1[1] = (rowLen & 0xFFFFu) << 16;             // tensor_dim0[15:0]
  g1[2] = (rowLen >> 16) | ((rows & 0xFFFFu) << 16);   // dim0[31:16] | dim1[15:0]
  g1[3] = (rows >> 16) | (32u << 16);           // dim1[31:16] | tile_dim0=32
  g1[4] = 128u;                                 // tile_dim1=128, tile_dim2=0
  g1[5] = rowLen;                               // tensor_dim0_stride[31:0]
  g1[6] = 0u;                                   // stride0[47:32] | stride1 lo
  g1[7] = 0u;
  v4u g2 = {0u, 0u, 0u, 0u};
  v4u g3 = {0u, 0u, 0u, 0u};
  asm volatile("tensor_load_to_lds %0, %1, %2, %3"
               :: "s"(g0), "s"(g1), "s"(g2), "s"(g3)
               : "memory");
}

// ---------------------------------------------------------------------------
// f32 -> bf16 bulk convert (4 elems/thread)
// ---------------------------------------------------------------------------
__global__ __launch_bounds__(256) void cvt_bf16_kernel(const float* __restrict__ src,
                                                       bf16* __restrict__ dst, int n4) {
  int i = blockIdx.x * blockDim.x + threadIdx.x;
  if (i >= n4) return;
  v4f f = *(const v4f*)(src + (size_t)i * 4);
  v4bf o;
  o[0] = f2bf(f[0]); o[1] = f2bf(f[1]); o[2] = f2bf(f[2]); o[3] = f2bf(f[3]);
  *(v4bf*)(dst + (size_t)i * 4) = o;
}

// ---------------------------------------------------------------------------
// f32 (K x N) -> bf16 transposed (N x K): weights become row-major [N][K]
// so GEMM B-tiles are straight TDM copies (no LDS transpose needed).
// ---------------------------------------------------------------------------
__global__ __launch_bounds__(256) void cvt_bf16_t_kernel(const float* __restrict__ src,
                                                         bf16* __restrict__ dst,
                                                         int srcRows, int srcCols) {
  size_t idx = (size_t)blockIdx.x * blockDim.x + threadIdx.x;
  size_t total = (size_t)srcRows * srcCols;
  if (idx >= total) return;
  int n = (int)(idx / srcRows);   // dst row  = src col
  int k = (int)(idx % srcRows);   // dst col  = src row
  dst[idx] = f2bf(src[(size_t)k * srcCols + n]);
}

// ---------------------------------------------------------------------------
// C(f32, MxN) = A(bf16, MxK row-major) * Bt(bf16, NxK row-major)^T
// 128x128 block tile, BK=32, 8 waves (2x4), each wave 64x32 (4x2 WMMA accs).
// Tiles streamed into double-buffered LDS by the Tensor Data Mover (wave 0),
// overlapped with WMMA compute; completion via s_wait_tensorcnt.
// M,N % 128 == 0, K % 32 == 0 (true for all uses here).
// ---------------------------------------------------------------------------
__global__ __launch_bounds__(256) void gemm_bf16_tdm_kernel(const bf16* __restrict__ A,
                                                            const bf16* __restrict__ Bt,
                                                            float* __restrict__ C,
                                                            int M, int N, int K) {
  __shared__ bf16 As[2][128][32];   // 2 x 8 KB
  __shared__ bf16 Bs[2][128][32];   // 2 x 8 KB
  const int tid  = threadIdx.x;
  const int wid  = tid >> 5;
  const int lane = tid & 31;
  const int lm = lane & 15;
  const int lh = lane >> 4;
  const int kb = lh * 8;
  const int wm = (wid >> 2) * 64;
  const int wn = (wid & 3) * 32;
  const int m0 = blockIdx.y * 128;
  const int n0 = blockIdx.x * 128;

  v8f acc[4][2];
#pragma unroll
  for (int i = 0; i < 4; ++i)
#pragma unroll
    for (int j = 0; j < 2; ++j) acc[i][j] = v8f_zero();

  const bf16* aT = A  + (size_t)m0 * K;
  const bf16* bT = Bt + (size_t)n0 * K;
  const int nIter = K / 32;

  if (wid == 0) {                       // prologue: fill buffer 0
    tdm_load_tile_2d(aT, lds_off(&As[0][0][0]), (unsigned)K, (unsigned)(M - m0));
    tdm_load_tile_2d(bT, lds_off(&Bs[0][0][0]), (unsigned)K, (unsigned)(N - n0));
  }

  for (int it = 0; it < nIter; ++it) {
    if (wid == 0) __builtin_amdgcn_s_wait_tensorcnt(0);   // current tile landed
    __syncthreads();   // publishes TDM data; seals prev iter's LDS reads
    if (wid == 0 && it + 1 < nIter) {   // prefetch next tile into other buffer
      const int k1 = (it + 1) * 32;
      const int nb = (it + 1) & 1;
      tdm_load_tile_2d(aT + k1, lds_off(&As[nb][0][0]), (unsigned)K, (unsigned)(M - m0));
      tdm_load_tile_2d(bT + k1, lds_off(&Bs[nb][0][0]), (unsigned)K, (unsigned)(N - n0));
    }
    const int cb = it & 1;
    v16bf af[4], bfr[2];
#pragma unroll
    for (int i = 0; i < 4; ++i)
      af[i] = make_frag(&As[cb][wm + i * 16 + lm][kb], &As[cb][wm + i * 16 + lm][16 + kb]);
#pragma unroll
    for (int j = 0; j < 2; ++j)
      bfr[j] = make_frag(&Bs[cb][wn + j * 16 + lm][kb], &Bs[cb][wn + j * 16 + lm][16 + kb]);
#pragma unroll
    for (int i = 0; i < 4; ++i)
#pragma unroll
      for (int j = 0; j < 2; ++j)
        acc[i][j] = WMMA_BF16(af[i], bfr[j], acc[i][j]);
  }

  // C/D layout: lane holds (m = r + 8*lh, n = lm) in VGPR r.
#pragma unroll
  for (int i = 0; i < 4; ++i)
#pragma unroll
    for (int j = 0; j < 2; ++j)
#pragma unroll
      for (int r = 0; r < 8; ++r)
        C[(size_t)(m0 + wm + i * 16 + r + 8 * lh) * N + (n0 + wn + j * 16 + lm)] =
            acc[i][j][r];
}

// ---------------------------------------------------------------------------
// RoPE + relayout: src f32 (b, s, nh, 64) -> dst bf16 (b, nh, s, 64)
// ---------------------------------------------------------------------------
__global__ __launch_bounds__(256) void rope_pack_kernel(const float* __restrict__ src,
                                                        const float* __restrict__ cosb,
                                                        const float* __restrict__ sinb,
                                                        bf16* __restrict__ dst, int nheads) {
  size_t idx = (size_t)blockIdx.x * blockDim.x + threadIdx.x;
  size_t total = (size_t)kBsz * kSeq * nheads * (kHD / 2);
  if (idx >= total) return;
  int p = (int)(idx & (kHD / 2 - 1));
  int h = (int)((idx >> 5) % nheads);
  int s = (int)((idx / ((size_t)(kHD / 2) * nheads)) % kSeq);
  int b = (int)(idx / ((size_t)(kHD / 2) * nheads * kSeq));
  const float* sp = src + (((size_t)b * kSeq + s) * nheads + h) * kHD + 2 * p;
  float tr = sp[0], ti = sp[1];
  float c  = cosb[s * (kHD / 2) + p];
  float sn = sinb[s * (kHD / 2) + p];
  bf16* d = dst + (((size_t)b * nheads + h) * kSeq + s) * kHD + 2 * p;
  d[0] = f2bf(tr * c - ti * sn);
  d[1] = f2bf(tr * sn + ti * c);
}

// ---------------------------------------------------------------------------
// V relayout: src f32 (b, s, kv, 64) -> dst bf16 (b, kv, 64, s)   (V^T)
// ---------------------------------------------------------------------------
__global__ __launch_bounds__(256) void vtrans_kernel(const float* __restrict__ src,
                                                     bf16* __restrict__ dst) {
  size_t idx = (size_t)blockIdx.x * blockDim.x + threadIdx.x;
  size_t total = (size_t)kBsz * kSeq * kNKV * kHD;
  if (idx >= total) return;
  int d_ = (int)(idx & (kHD - 1));
  int kv = (int)((idx >> 6) & (kNKV - 1));
  int s  = (int)((idx >> 9) & (kSeq - 1));
  int b  = (int)(idx >> 20);
  dst[(((size_t)b * kNKV + kv) * kHD + d_) * kSeq + s] =
      f2bf(src[(((size_t)b * kSeq + s) * kNKV + kv) * kHD + d_]);
}

// ---------------------------------------------------------------------------
// Flash attention (causal, GQA 4:1). 1 wave = 16 query rows, 32-key blocks.
// Q  bf16 (b, 32, s, 64)  row-major
// K  bf16 (b, 8,  s, 64)  row-major  (B-fragment cols of K^T = K rows)
// Vt bf16 (b, 8, 64, s)   row-major  (B-fragment cols of V = Vt rows)
// O  bf16 (b, s, 2048)
// ---------------------------------------------------------------------------
__global__ __launch_bounds__(128) void flash_attn_kernel(const bf16* __restrict__ Q,
                                                         const bf16* __restrict__ Kc,
                                                         const bf16* __restrict__ Vt,
                                                         bf16* __restrict__ O) {
  __shared__ bf16 Plds[4][16 * 32];  // per-wave P staging (C-layout -> A-layout)
  const int tid = threadIdx.x;
  const int wid = tid >> 5, lane = tid & 31;
  const int lm = lane & 15, lh = lane >> 4, kb = lh * 8;
  const int qbase = (blockIdx.x * 4 + wid) * 16;
  const int bh = blockIdx.y;
  const int b = bh >> 5, h = bh & 31, kv = h >> 2;

  const bf16* q  = Q  + ((size_t)(b * kNH + h) * kSeq + qbase) * kHD;
  const bf16* k  = Kc + ((size_t)(b * kNKV + kv) * kSeq) * kHD;
  const bf16* vt = Vt + ((size_t)(b * kNKV + kv) * kHD) * (size_t)kSeq;

  const v16bf aq0 = make_frag(q + lm * kHD + kb,      q + lm * kHD + 16 + kb);
  const v16bf aq1 = make_frag(q + lm * kHD + 32 + kb, q + lm * kHD + 48 + kb);

  v8f o[4];
#pragma unroll
  for (int dt = 0; dt < 4; ++dt) o[dt] = v8f_zero();
  float rmax[8], rsum[8];
#pragma unroll
  for (int r = 0; r < 8; ++r) { rmax[r] = -1e30f; rsum[r] = 0.0f; }
  const float scale = 0.125f;  // 1/sqrt(64)

  const int qend = qbase + 15;
  for (int j0 = 0; j0 <= qend; j0 += 32) {
    // ---- S = Q K^T (two 16x16 tiles covering 32 keys) ----
    float sv[2][8];
#pragma unroll
    for (int t = 0; t < 2; ++t) {
      const bf16* kr = k + (size_t)(j0 + t * 16 + lm) * kHD;
      v16bf bk0 = make_frag(kr + kb,      kr + 16 + kb);
      v16bf bk1 = make_frag(kr + 32 + kb, kr + 48 + kb);
      v8f s = v8f_zero();
      s = WMMA_BF16(aq0, bk0, s);
      s = WMMA_BF16(aq1, bk1, s);
      const int key = j0 + t * 16 + lm;
#pragma unroll
      for (int r = 0; r < 8; ++r) {
        const int qr = qbase + r + 8 * lh;
        float v = s[r] * scale;
        sv[t][r] = (key > qr) ? -1e30f : v;  // causal mask, branchless
      }
    }
    // ---- online softmax over this 32-key block ----
#pragma unroll
    for (int r = 0; r < 8; ++r) {
      float mx = fmaxf(sv[0][r], sv[1][r]);
#pragma unroll
      for (int off = 1; off < 16; off <<= 1)
        mx = fmaxf(mx, __shfl_xor(mx, off, 32));   // reduce within 16-lane half
      float nm   = fmaxf(rmax[r], mx);
      float corr = __expf(rmax[r] - nm);
      rmax[r] = nm;
      float p0 = __expf(sv[0][r] - nm);
      float p1 = __expf(sv[1][r] - nm);
      float ps = p0 + p1;
#pragma unroll
      for (int off = 1; off < 16; off <<= 1)
        ps += __shfl_xor(ps, off, 32);
      rsum[r] = rsum[r] * corr + ps;
#pragma unroll
      for (int dt = 0; dt < 4; ++dt) o[dt][r] *= corr;
      const int m = r + 8 * lh;
      Plds[wid][m * 32 + lm]      = f2bf(p0);
      Plds[wid][m * 32 + 16 + lm] = f2bf(p1);
    }
    // Wave-private LDS region; only need this wave's DS stores complete.
    asm volatile("s_wait_dscnt 0" ::: "memory");

    // ---- O += P V (P as A-frag from LDS, V columns contiguous via V^T) ----
    v16bf pa = make_frag(&Plds[wid][lm * 32 + kb], &Plds[wid][lm * 32 + 16 + kb]);
#pragma unroll
    for (int dt = 0; dt < 4; ++dt) {
      const bf16* vr = vt + (size_t)(dt * 16 + lm) * kSeq + j0;
      v16bf bv = make_frag(vr + kb, vr + 16 + kb);
      o[dt] = WMMA_BF16(pa, bv, o[dt]);
    }
  }

  float inv[8];
#pragma unroll
  for (int r = 0; r < 8; ++r) inv[r] = 1.0f / rsum[r];
  const size_t obase = ((size_t)b * kSeq + qbase) * kDim + (size_t)h * kHD;
#pragma unroll
  for (int dt = 0; dt < 4; ++dt)
#pragma unroll
    for (int r = 0; r < 8; ++r)
      O[obase + (size_t)(r + 8 * lh) * kDim + dt * 16 + lm] = f2bf(o[dt][r] * inv[r]);
}

// ---------------------------------------------------------------------------
// Host launcher
// ---------------------------------------------------------------------------
extern "C" void kernel_launch(void* const* d_in, const int* in_sizes, int n_in,
                              void* d_out, int out_size, void* d_ws, size_t ws_size,
                              hipStream_t stream) {
  (void)in_sizes; (void)n_in; (void)out_size; (void)ws_size;
  const float* x    = (const float*)d_in[0];
  const float* fcos = (const float*)d_in[1];
  const float* fsin = (const float*)d_in[2];
  const float* wq   = (const float*)d_in[3];
  const float* wk   = (const float*)d_in[4];
  const float* wv   = (const float*)d_in[5];
  const float* wo   = (const float*)d_in[6];
  float* out = (float*)d_out;

  char* p = (char*)d_ws;
  auto take = [&](size_t bytes) -> char* {
    char* r = p;
    p += (bytes + 255) & ~(size_t)255;
    return r;
  };
  bf16*  xb  = (bf16*)take((size_t)kRows * kDim * 2);
  bf16*  wqb = (bf16*)take((size_t)kDim * kDim * 2);   // transposed: (N=2048, K=2048)
  bf16*  wkb = (bf16*)take((size_t)kDim * kKVD * 2);   // transposed: (N=512,  K=2048)
  bf16*  wvb = (bf16*)take((size_t)kDim * kKVD * 2);   // transposed
  bf16*  wob = (bf16*)take((size_t)kDim * kDim * 2);   // transposed
  float* Qf  = (float*)take((size_t)kRows * kDim * 4);
  float* Kf  = (float*)take((size_t)kRows * kKVD * 4);
  float* Vf  = (float*)take((size_t)kRows * kKVD * 4);
  bf16*  Qb  = (bf16*)take((size_t)kRows * kDim * 2);
  bf16*  Kb  = (bf16*)take((size_t)kRows * kKVD * 2);
  bf16*  Vtb = (bf16*)take((size_t)kRows * kKVD * 2);
  bf16*  AOb = (bf16*)Qf;  // reuse Qf region: free after RoPE-Q consumed it

  {
    size_t n = (size_t)kRows * kDim;
    int n4 = (int)(n / 4);
    cvt_bf16_kernel<<<(n4 + 255) / 256, 256, 0, stream>>>(x, xb, n4);
  }
  auto cvtT = [&](const float* s, bf16* d, int srcRows, int srcCols) {
    size_t n = (size_t)srcRows * srcCols;
    cvt_bf16_t_kernel<<<(unsigned)((n + 255) / 256), 256, 0, stream>>>(s, d, srcRows, srcCols);
  };
  cvtT(wq, wqb, kDim, kDim);
  cvtT(wk, wkb, kDim, kKVD);
  cvtT(wv, wvb, kDim, kKVD);
  cvtT(wo, wob, kDim, kDim);

  dim3 blk(256);
  gemm_bf16_tdm_kernel<<<dim3(kDim / 128, kRows / 128), blk, 0, stream>>>(xb, wqb, Qf, kRows, kDim, kDim);
  gemm_bf16_tdm_kernel<<<dim3(kKVD / 128, kRows / 128), blk, 0, stream>>>(xb, wkb, Kf, kRows, kKVD, kDim);
  gemm_bf16_tdm_kernel<<<dim3(kKVD / 128, kRows / 128), blk, 0, stream>>>(xb, wvb, Vf, kRows, kKVD, kDim);

  {
    size_t nq = (size_t)kBsz * kSeq * kNH * (kHD / 2);
    rope_pack_kernel<<<(unsigned)((nq + 255) / 256), 256, 0, stream>>>(Qf, fcos, fsin, Qb, kNH);
    size_t nk = (size_t)kBsz * kSeq * kNKV * (kHD / 2);
    rope_pack_kernel<<<(unsigned)((nk + 255) / 256), 256, 0, stream>>>(Kf, fcos, fsin, Kb, kNKV);
    size_t nv = (size_t)kBsz * kSeq * kNKV * kHD;
    vtrans_kernel<<<(unsigned)((nv + 255) / 256), 256, 0, stream>>>(Vf, Vtb);
  }

  flash_attn_kernel<<<dim3(kSeq / 64, kBsz * kNH), dim3(128), 0, stream>>>(Qb, Kb, Vtb, AOb);

  gemm_bf16_tdm_kernel<<<dim3(kDim / 128, kRows / 128), blk, 0, stream>>>(AOb, wob, out, kRows, kDim, kDim);
}
